// MultiModalModel_22016002360055
// MI455X (gfx1250) — matrix operand order repaired and checked by reading, not compiled, not run on hardware
//
#include <hip/hip_runtime.h>
#include <hip/hip_bf16.h>
#include <math.h>

// Problem dims (fixed by reference)
#define I_DIM 256
#define T_DIM 256
#define L_DIM 32
#define E_DIM 128
#define HW    49
#define TL    (T_DIM * L_DIM)   // 8192

typedef __attribute__((ext_vector_type(16))) __bf16 v16bf;
typedef __attribute__((ext_vector_type(8)))  __bf16 v8bf;
typedef __attribute__((ext_vector_type(8)))  float  v8f;

union FragU { v16bf v; v8bf h[2]; };

// s_wait_asynccnt: builtin if declared, else raw SOPP
#if defined(__has_builtin)
#if __has_builtin(__builtin_amdgcn_s_wait_asynccnt)
#define WAIT_ASYNC(n) __builtin_amdgcn_s_wait_asynccnt(n)
#endif
#endif
#ifndef WAIT_ASYNC
#define WAIT_ASYNC(n) asm volatile("s_wait_asynccnt %0" ::"i"(n) : "memory")
#endif

// ---------------------------------------------------------------------------
// Stage 1a: image_features [I,E,H,W] f32  ->  A[p][i][e] bf16
// ---------------------------------------------------------------------------
__global__ void __launch_bounds__(256)
cvt_image_kernel(const float* __restrict__ img, __bf16* __restrict__ A) {
  int idx = blockIdx.x * 256 + threadIdx.x;      // 49*256*128 = 1,605,632 total
  int p = idx >> 15;                             // / (256*128)
  int r = idx & 32767;
  int i = r >> 7;
  int e = r & 127;
  A[idx] = (__bf16)img[i * (E_DIM * HW) + e * HW + p];
}

// ---------------------------------------------------------------------------
// Stage 1b: text_features [T,L,E] f32  ->  B[e][tl] bf16 (transposed)
// ---------------------------------------------------------------------------
__global__ void __launch_bounds__(256)
cvt_text_kernel(const float* __restrict__ txt, __bf16* __restrict__ B) {
  int idx = blockIdx.x * 256 + threadIdx.x;      // 128*8192 = 1,048,576 total
  int e  = idx >> 13;
  int tl = idx & 8191;
  B[idx] = (__bf16)txt[tl * E_DIM + e];
}

// ---------------------------------------------------------------------------
// Stage 2: R[i, tl] = max_p  sum_e A_p[i,e] * B[e,tl]
// Workgroup: 256 threads = 8 waves (2 x 4); tile 64(i) x 128(tl).
// B slab (128x128, 32KB LDS) loaded once; its 8 wave-fragments are then
// hoisted into registers for the entire p-loop.
// A slab (64x128, 16KB) double-buffered via async global->LDS copies.
// ---------------------------------------------------------------------------
#define LDA_PAD 136   // 128 + 8 bf16 pad -> breaks LDS bank conflicts
#define LDB_PAD 136
#define ATILE   (64 * LDA_PAD)

// Issue one A-tile (64 rows x 128 e, bf16) as 4 async b128 loads per thread.
__device__ __forceinline__ void issue_async_a_tile(const __bf16* gbase,
                                                   __bf16* ldsbuf, int tid) {
#pragma unroll
  for (int k = 0; k < 4; ++k) {
    int idx = tid + k * 256;                 // 0..1023 (16B chunks)
    int row = idx >> 4;                      // 64 rows, 16 chunks/row
    int c   = (idx & 15) << 3;               // bf16 col, multiple of 8
    unsigned goff  = (unsigned)((row * E_DIM + c) * 2);   // byte offset, 16B aligned
    unsigned laddr = (unsigned)(uintptr_t)&ldsbuf[row * LDA_PAD + c]; // AS3 offset
    asm volatile("global_load_async_to_lds_b128 %0, %1, %2 offset:0"
                 :: "v"(laddr), "v"(goff), "s"(gbase) : "memory");
  }
}

__global__ void __launch_bounds__(256)
match_gemm_kernel(const __bf16* __restrict__ A, const __bf16* __restrict__ B,
                  float* __restrict__ R) {
  __shared__ __attribute__((aligned(16))) __bf16 Alds[2 * ATILE];
  __shared__ __attribute__((aligned(16))) __bf16 Blds[128 * LDB_PAD];

  const int tid   = threadIdx.x;
  const int lane  = tid & 31;
  const int wave  = tid >> 5;
  const int waveM = wave >> 2;            // 0..1
  const int waveN = wave & 3;             // 0..3
  const int iBase = blockIdx.x * 64;      // 4 blocks
  const int nBase = blockIdx.y * 128;     // 64 blocks

  const __bf16* Ablk = A + iBase * E_DIM; // + p*(I*E) per spatial step

  // Prologue: kick off async copy of A tile p=0 into buffer 0
  issue_async_a_tile(Ablk, &Alds[0], tid);

  // --- load B tile [128 e][128 tl] into LDS (once) ---
#pragma unroll
  for (int k = 0; k < 8; ++k) {
    int idx = tid + k * 256;              // 0..2047 16B chunks
    int row = idx >> 4;                   // 16 x 8bf16 per row
    int c   = (idx & 15) << 3;
    *(v8bf*)&Blds[row * LDB_PAD + c] = *(const v8bf*)&B[row * TL + nBase + c];
  }
  __syncthreads();                        // B slab visible to all waves

  // --- hoist this wave's B fragments into registers for the whole p-loop ---
  // B fragment layout: N = lane&15, K = {0..7,16..23} (lanes 0-15) / {8..15,24..31} (lanes 16-31).
  const int bN0 = waveN * 32;
  FragU fb[4][2];                         // [e-chunk][n-tile] -> 64 VGPRs
#pragma unroll
  for (int ec = 0; ec < 4; ++ec) {
    const int bK = ec * 32 + ((lane >> 4) << 3);
    const int bCol = bN0 + (lane & 15);
#pragma unroll
    for (int nt = 0; nt < 2; ++nt) {
#pragma unroll
      for (int s = 0; s < 8; ++s) {
        fb[ec][nt].v[s]     = Blds[(bK + s) * LDB_PAD + bCol + nt * 16];
        fb[ec][nt].v[s + 8] = Blds[(bK + 16 + s) * LDB_PAD + bCol + nt * 16];
      }
    }
  }

  v8f runmax[2][2];
#pragma unroll
  for (int m = 0; m < 2; ++m)
#pragma unroll
    for (int n = 0; n < 2; ++n)
#pragma unroll
      for (int j = 0; j < 8; ++j) runmax[m][n][j] = -3.402823466e38f;

  // A fragment addressing (ISA 16-bit A 16x32 layout):
  //   lanes 0-15: row = lane,  K = e0 + {0..7, 16..23}
  //   lanes 16-31: row = lane-16, K = e0 + {8..15, 24..31}
  const int aRow0 = waveM * 32 + (lane & 15);
  const int aRow1 = aRow0 + 16;
  const int aSel  = (lane >> 4) << 3;     // 0 or 8

  for (int p = 0; p < HW; ++p) {
    // Prefetch next spatial slab into the alternate buffer (overlaps compute)
    if (p + 1 < HW)
      issue_async_a_tile(Ablk + (p + 1) * (I_DIM * E_DIM),
                         &Alds[((p + 1) & 1) * ATILE], tid);

    // Wait for tile p's 4 async loads (in-order completion per wave)
    if (p + 1 < HW) { WAIT_ASYNC(4); } else { WAIT_ASYNC(0); }
    __syncthreads();                      // all waves' tile-p data visible

    const __bf16* Acur = &Alds[(p & 1) * ATILE];

    v8f acc[2][2];
#pragma unroll
    for (int m = 0; m < 2; ++m)
#pragma unroll
      for (int n = 0; n < 2; ++n)
#pragma unroll
        for (int j = 0; j < 8; ++j) acc[m][n][j] = 0.0f;

#pragma unroll
    for (int ec = 0; ec < 4; ++ec) {
      const int e0 = ec * 32;
      FragU fa0, fa1;
      fa0.h[0] = *(const v8bf*)&Acur[aRow0 * LDA_PAD + e0 + aSel];
      fa0.h[1] = *(const v8bf*)&Acur[aRow0 * LDA_PAD + e0 + 16 + aSel];
      fa1.h[0] = *(const v8bf*)&Acur[aRow1 * LDA_PAD + e0 + aSel];
      fa1.h[1] = *(const v8bf*)&Acur[aRow1 * LDA_PAD + e0 + 16 + aSel];

      acc[0][0] = __builtin_amdgcn_wmma_f32_16x16x32_bf16(
          false, fa0.v, false, fb[ec][0].v, (short)0, acc[0][0], false, false);
      acc[0][1] = __builtin_amdgcn_wmma_f32_16x16x32_bf16(
          false, fa0.v, false, fb[ec][1].v, (short)0, acc[0][1], false, false);
      acc[1][0] = __builtin_amdgcn_wmma_f32_16x16x32_bf16(
          false, fa1.v, false, fb[ec][0].v, (short)0, acc[1][0], false, false);
      acc[1][1] = __builtin_amdgcn_wmma_f32_16x16x32_bf16(
          false, fa1.v, false, fb[ec][1].v, (short)0, acc[1][1], false, false);
    }

    // Running max via llvm.maxnum: compiler handles the WMMA->VALU hazard NOPs
#pragma unroll
    for (int m = 0; m < 2; ++m)
#pragma unroll
      for (int n = 0; n < 2; ++n)
        runmax[m][n] = __builtin_elementwise_max(runmax[m][n], acc[m][n]);

    __syncthreads();  // all waves done reading buf[p&1] before it is re-filled
  }

  // --- write R (f32 C/D layout: N = lane&15, M = vgpr + 8*(lane>=16)) ---
  const int cRowOff = (lane >> 4) << 3;
  const int cCol    = lane & 15;
#pragma unroll
  for (int m = 0; m < 2; ++m)
#pragma unroll
    for (int n = 0; n < 2; ++n)
#pragma unroll
      for (int r = 0; r < 8; ++r) {
        int gi = iBase + waveM * 32 + m * 16 + r + cRowOff;
        int gn = nBase + waveN * 32 + n * 16 + cCol;
        R[gi * TL + gn] = runmax[m][n][r];
      }
}

// ---------------------------------------------------------------------------
// Stage 3: out_img[i,t] = scale/len[t] * sum_l R[i, t*32+l]; out_txt = T(out_img)
// ---------------------------------------------------------------------------
__global__ void __launch_bounds__(256)
reduce_kernel(const float* __restrict__ R, const int* __restrict__ len,
              const float* __restrict__ nlt, float* __restrict__ out) {
  int idx = blockIdx.x * 256 + threadIdx.x;  // 65536 = I*T
  int i = idx >> 8;
  int t = idx & 255;
  const float* base = R + i * TL + t * L_DIM;
  float s = 0.0f;
#pragma unroll
  for (int l = 0; l < L_DIM; ++l) s += base[l];
  float val = s * expf(nlt[0]) / (float)len[t];
  out[i * T_DIM + t] = val;                       // logits_per_image
  out[I_DIM * T_DIM + t * I_DIM + i] = val;       // logits_per_text
}

// ---------------------------------------------------------------------------
extern "C" void kernel_launch(void* const* d_in, const int* in_sizes, int n_in,
                              void* d_out, int out_size, void* d_ws, size_t ws_size,
                              hipStream_t stream) {
  (void)in_sizes; (void)n_in; (void)out_size; (void)ws_size;
  const float* img = (const float*)d_in[0];
  const float* txt = (const float*)d_in[1];
  const int*   len = (const int*)d_in[2];
  const float* nlt = (const float*)d_in[3];

  // Workspace layout (all 256B-aligned offsets), total ~13.7 MB:
  //   A bf16: 49*256*128*2 = 3,211,264 B
  //   B bf16: 128*8192*2   = 2,097,152 B
  //   R f32 : 256*8192*4   = 8,388,608 B
  char* ws = (char*)d_ws;
  __bf16* Abf = (__bf16*)(ws);
  __bf16* Bbf = (__bf16*)(ws + 3211264);
  float*  R   = (float*)(ws + 5308416);

  cvt_image_kernel<<<6272, 256, 0, stream>>>(img, Abf);
  cvt_text_kernel<<<4096, 256, 0, stream>>>(txt, Bbf);
  dim3 grid(4, 64);
  match_gemm_kernel<<<grid, 256, 0, stream>>>(Abf, Bbf, R);
  reduce_kernel<<<256, 256, 0, stream>>>(R, len, nlt, (float*)d_out);
}
